// UnquantizedMoELayer_67826123538954
// MI455X (gfx1250) — matrix operand order repaired and checked by reading, not compiled, not run on hardware
//
#include <hip/hip_runtime.h>
#include <hip/hip_bf16.h>
#include <math.h>

// Problem constants
#define M_TOK 2048
#define N_EXP 8
#define K_DIM 2048
#define N_INT 5632
#define TOPK  2
#define CAP   (M_TOK * TOPK)   // worst-case tokens per expert (dup topk ids possible)

// Tile config: 256 threads = 8 waves (wave32). Block tile 128(M) x 64(N), K-step 32.
// Waves in 4x2 grid; each wave owns 32x32 = 2x2 fragments of 16x16.
#define BM 128
#define BN 64
#define BK 32
#define LDST 40   // LDS row stride in bf16 elements (80B: 16B-aligned, conflict-free-ish)

typedef __bf16 bf16;
typedef __attribute__((ext_vector_type(16))) __bf16 v16bf;
typedef __attribute__((ext_vector_type(8)))  __bf16 v8bf;
typedef __attribute__((ext_vector_type(4)))  __bf16 v4bf;
typedef __attribute__((ext_vector_type(8)))  float  v8f;
typedef __attribute__((ext_vector_type(4)))  float  v4f;
typedef __attribute__((ext_vector_type(4)))  int    v4i;

// ---- CDNA5 async global->LDS copy (ASYNCcnt path), guarded so build can't break ----
#if defined(__has_builtin)
#if __has_builtin(__builtin_amdgcn_global_load_async_to_lds_b128) && \
    __has_builtin(__builtin_amdgcn_s_wait_asynccnt)
#define USE_ASYNC_LDS 1
#endif
#endif
#ifndef USE_ASYNC_LDS
#define USE_ASYNC_LDS 0
#endif

typedef __attribute__((address_space(1))) v4i* gptr_v4i;   // global int4*
typedef __attribute__((address_space(3))) v4i* lptr_v4i;   // LDS int4*

__device__ __forceinline__ void cp_async_b128(bf16* lds_dst, const bf16* g_src) {
#if USE_ASYNC_LDS
  gptr_v4i gp = (gptr_v4i)(const_cast<bf16*>(g_src));
  lptr_v4i lp = (lptr_v4i)(lds_dst);
  __builtin_amdgcn_global_load_async_to_lds_b128(gp, lp, 0, 0);
#else
  *(v8bf*)lds_dst = *(const v8bf*)g_src;
#endif
}

__device__ __forceinline__ void async_fence() {
#if USE_ASYNC_LDS
  __builtin_amdgcn_s_wait_asynccnt(0);
#endif
}

__device__ __forceinline__ v8f wmma_bf16(v16bf a, v16bf b, v8f c) {
  // (neg_a, A, neg_b, B, c_mod, C, reuse_a, reuse_b)
  return __builtin_amdgcn_wmma_f32_16x16x32_bf16(false, a, false, b, (short)0, c, false, false);
}

__device__ __forceinline__ v16bf cat8(v8bf lo, v8bf hi) {
  return __builtin_shufflevector(lo, hi, 0,1,2,3,4,5,6,7,8,9,10,11,12,13,14,15);
}

__device__ __forceinline__ v4bf cvt4(v4f v) {
  v4bf r = { (__bf16)v[0], (__bf16)v[1], (__bf16)v[2], (__bf16)v[3] };
  return r;
}

// ---------------- utility kernels ----------------

__global__ void zero_f32_kernel(float* __restrict__ p, int n) {
  int i = blockIdx.x * blockDim.x + threadIdx.x;
  if (i < n) p[i] = 0.0f;
}

__global__ void route_kernel(const int* __restrict__ ids, const float* __restrict__ w,
                             int* __restrict__ counts, int* __restrict__ perm,
                             float* __restrict__ wgt) {
  int i = blockIdx.x * blockDim.x + threadIdx.x;
  if (i >= M_TOK * TOPK) return;
  int e = ids[i];
  int slot = atomicAdd(&counts[e], 1);
  perm[e * CAP + slot] = i / TOPK;  // token row
  wgt[e * CAP + slot]  = w[i];
}

// ---------------- GEMM1: h = silu(x W1g^T) * (x W1u^T), gathered rows ----------------

__global__ __launch_bounds__(256)
void moe_gemm1(const float* __restrict__ x,      // (M, K)
               const float* __restrict__ w1,     // (2N, K) this expert
               const int*   __restrict__ counts,
               const int*   __restrict__ perm,   // this expert's row list
               bf16*        __restrict__ H,      // (CAP, N) bf16 scratch
               int e)
{
  const int cnt = counts[e];
  const int rowBase = blockIdx.x * BM;
  if (rowBase >= cnt) return;
  const int colBase = blockIdx.y * BN;   // intermediate column tile

  const int tid  = threadIdx.x;
  const int lane = tid & 31;
  const int wv   = tid >> 5;
  const int wr   = (wv & 3) * 32;   // wave row offset in tile
  const int wc   = (wv >> 2) * 32;  // wave col offset in tile
  const int lh   = lane >> 4;       // lane half (0/1)
  const int ll   = lane & 15;

  __shared__ bf16 sA [BM * LDST];
  __shared__ bf16 sBg[BN * LDST];
  __shared__ bf16 sBu[BN * LDST];
  __shared__ int  sPerm[BM];

  if (tid < BM) {
    int gr = rowBase + tid;
    sPerm[tid] = (gr < cnt) ? perm[gr] : 0;   // clamp: OOB rows read token 0, discarded later
  }
  __syncthreads();

  v8f accg[2][2] = {};
  v8f accu[2][2] = {};

  const int c4 = tid & 7;    // float4 column group within 32-wide K slab
  const int r0 = tid >> 3;   // 0..31

  // Hoist loop-invariant gather base pointers out of the k-loop
  // (LDS reads can't be hoisted past barriers by the compiler).
  const float* aptr[4];
  #pragma unroll
  for (int i = 0; i < 4; ++i)
    aptr[i] = x + (size_t)sPerm[r0 + 32 * i] * K_DIM + c4 * 4;
  const float* bgptr[2];
  const float* buptr[2];
  #pragma unroll
  for (int i = 0; i < 2; ++i) {
    int r = r0 + 32 * i;  // 0..63
    bgptr[i] = w1 + (size_t)(colBase + r) * K_DIM + c4 * 4;
    buptr[i] = w1 + (size_t)(N_INT + colBase + r) * K_DIM + c4 * 4;
  }

  for (int k0 = 0; k0 < K_DIM; k0 += BK) {
    // ---- stage A tile (gathered x rows, f32 -> bf16) ----
    #pragma unroll
    for (int i = 0; i < 4; ++i) {
      int r = r0 + 32 * i;
      v4f v = *(const v4f*)(aptr[i] + k0);
      *(v4bf*)&sA[r * LDST + c4 * 4] = cvt4(v);
    }
    // ---- stage B tiles: gate rows [colBase..) and up rows [N+colBase..) ----
    #pragma unroll
    for (int i = 0; i < 2; ++i) {
      int r = r0 + 32 * i;
      const float* pg = bgptr[i] + k0;
      const float* pu = buptr[i] + k0;
      if (k0 + BK < K_DIM) {
        __builtin_prefetch((const void*)(pg + BK), 0, 3);
        __builtin_prefetch((const void*)(pu + BK), 0, 3);
      }
      v4f g = *(const v4f*)pg;
      v4f u = *(const v4f*)pu;
      *(v4bf*)&sBg[r * LDST + c4 * 4] = cvt4(g);
      *(v4bf*)&sBu[r * LDST + c4 * 4] = cvt4(u);
    }
    __syncthreads();

    // ---- fragments ----
    v16bf af[2];
    #pragma unroll
    for (int rf = 0; rf < 2; ++rf) {
      const bf16* p = &sA[(wr + rf * 16 + ll) * LDST + lh * 8];
      af[rf] = cat8(*(const v8bf*)p, *(const v8bf*)(p + 16));
    }
    v16bf bg[2], bu[2];
    #pragma unroll
    for (int cf = 0; cf < 2; ++cf) {
      const bf16* pg = &sBg[(wc + cf * 16 + ll) * LDST + lh * 8];
      const bf16* pu = &sBu[(wc + cf * 16 + ll) * LDST + lh * 8];
      bg[cf] = cat8(*(const v8bf*)pg, *(const v8bf*)(pg + 16));
      bu[cf] = cat8(*(const v8bf*)pu, *(const v8bf*)(pu + 16));
    }
    #pragma unroll
    for (int rf = 0; rf < 2; ++rf)
      #pragma unroll
      for (int cf = 0; cf < 2; ++cf) {
        accg[rf][cf] = wmma_bf16(af[rf], bg[cf], accg[rf][cf]);
        accu[rf][cf] = wmma_bf16(af[rf], bu[cf], accu[rf][cf]);
      }
    __syncthreads();
  }

  // ---- epilogue: silu(g)*u -> H (bf16). C/D layout: VGPR v -> M = lh*8+v, N = ll ----
  #pragma unroll
  for (int rf = 0; rf < 2; ++rf)
    #pragma unroll
    for (int cf = 0; cf < 2; ++cf)
      #pragma unroll
      for (int v = 0; v < 8; ++v) {
        int ml = wr + rf * 16 + lh * 8 + v;
        int gr = rowBase + ml;
        if (gr < cnt) {
          float g = accg[rf][cf][v];
          float u = accu[rf][cf][v];
          float h = (g / (1.0f + __expf(-g))) * u;
          H[(size_t)gr * N_INT + colBase + wc + cf * 16 + ll] = (bf16)h;
        }
      }
}

// ---------------- GEMM2: out[perm[r]] += w * (H W2^T) ----------------

__global__ __launch_bounds__(256)
void moe_gemm2(const bf16*  __restrict__ H,      // (CAP, N) bf16
               const float* __restrict__ w2,     // (K, N) this expert
               const int*   __restrict__ counts,
               const int*   __restrict__ perm,
               const float* __restrict__ wgt,
               float*       __restrict__ out,    // (M, K)
               int e)
{
  const int cnt = counts[e];
  const int rowBase = blockIdx.x * BM;
  if (rowBase >= cnt) return;
  const int kBase = blockIdx.y * BN;   // output K-column tile

  const int tid  = threadIdx.x;
  const int lane = tid & 31;
  const int wv   = tid >> 5;
  const int wr   = (wv & 3) * 32;
  const int wc   = (wv >> 2) * 32;
  const int lh   = lane >> 4;
  const int ll   = lane & 15;

  __shared__ bf16 sA[BM * LDST];
  __shared__ bf16 sB[BN * LDST];
  __shared__ int   sPerm[BM];
  __shared__ float sW[BM];

  if (tid < BM) {
    int gr = rowBase + tid;
    sPerm[tid] = (gr < cnt) ? perm[gr] : 0;
    sW[tid]    = (gr < cnt) ? wgt[gr] : 0.0f;
  }
  __syncthreads();

  v8f acc[2][2] = {};

  const int c8  = tid & 3;   // 8-elem bf16 group within 32-wide slab
  const int r0a = tid >> 2;  // 0..63
  const int c4  = tid & 7;
  const int r0  = tid >> 3;  // 0..31

  // Hoisted, clamped base pointers (OOB rows -> row 0, results discarded in epilogue)
  const bf16* hptr[2];
  #pragma unroll
  for (int i = 0; i < 2; ++i) {
    int gr = rowBase + r0a + 64 * i;
    int hr = (gr < cnt) ? gr : 0;
    hptr[i] = H + (size_t)hr * N_INT + c8 * 8;
  }
  const float* bptr[2];
  #pragma unroll
  for (int i = 0; i < 2; ++i)
    bptr[i] = w2 + (size_t)(kBase + r0 + 32 * i) * N_INT + c4 * 4;

  for (int n0 = 0; n0 < N_INT; n0 += BK) {
    // ---- stage A tile from H (bf16 raw copy): CDNA5 async global->LDS ----
    #pragma unroll
    for (int i = 0; i < 2; ++i) {
      int r = r0a + 64 * i;
      cp_async_b128(&sA[r * LDST + c8 * 8], hptr[i] + n0);
    }
    // ---- stage B tile: w2 rows kBase..kBase+63, cols n0..n0+31 (f32 -> bf16) ----
    #pragma unroll
    for (int i = 0; i < 2; ++i) {
      int r = r0 + 32 * i;
      const float* pb = bptr[i] + n0;
      if (n0 + BK < N_INT) __builtin_prefetch((const void*)(pb + BK), 0, 3);
      v4f v = *(const v4f*)pb;
      *(v4bf*)&sB[r * LDST + c4 * 4] = cvt4(v);
    }
    async_fence();        // s_wait_asynccnt 0 before the barrier
    __syncthreads();

    v16bf af[2];
    #pragma unroll
    for (int rf = 0; rf < 2; ++rf) {
      const bf16* p = &sA[(wr + rf * 16 + ll) * LDST + lh * 8];
      af[rf] = cat8(*(const v8bf*)p, *(const v8bf*)(p + 16));
    }
    v16bf bf_[2];
    #pragma unroll
    for (int cf = 0; cf < 2; ++cf) {
      const bf16* p = &sB[(wc + cf * 16 + ll) * LDST + lh * 8];
      bf_[cf] = cat8(*(const v8bf*)p, *(const v8bf*)(p + 16));
    }
    #pragma unroll
    for (int rf = 0; rf < 2; ++rf)
      #pragma unroll
      for (int cf = 0; cf < 2; ++cf)
        acc[rf][cf] = wmma_bf16(af[rf], bf_[cf], acc[rf][cf]);
    __syncthreads();
  }

  // ---- scatter-accumulate epilogue (atomic: dup topk ids may repeat a row) ----
  #pragma unroll
  for (int rf = 0; rf < 2; ++rf)
    #pragma unroll
    for (int cf = 0; cf < 2; ++cf)
      #pragma unroll
      for (int v = 0; v < 8; ++v) {
        int ml = wr + rf * 16 + lh * 8 + v;
        int gr = rowBase + ml;
        if (gr < cnt) {
          float val = sW[ml] * acc[rf][cf][v];
          atomicAdd(&out[(size_t)sPerm[ml] * K_DIM + kBase + wc + cf * 16 + ll], val);
        }
      }
}

// ---------------- launch ----------------

extern "C" void kernel_launch(void* const* d_in, const int* in_sizes, int n_in,
                              void* d_out, int out_size, void* d_ws, size_t ws_size,
                              hipStream_t stream) {
  const float* x   = (const float*)d_in[0];
  const float* gup = (const float*)d_in[1];   // (E, 2N, K)
  const float* dpj = (const float*)d_in[2];   // (E, K, N)
  const float* tkw = (const float*)d_in[3];   // (M, TOPK)
  const int*   tki = (const int*)  d_in[4];   // (M, TOPK)
  float* out = (float*)d_out;

  // workspace layout
  int*   counts = (int*)d_ws;                     // 64 ints (8 used)
  int*   perm   = counts + 64;                    // E*CAP ints
  float* wgt    = (float*)(perm + N_EXP * CAP);   // E*CAP floats
  bf16*  H      = (bf16*)(wgt + N_EXP * CAP);     // CAP*N bf16 (~46 MB)

  zero_f32_kernel<<<(M_TOK * K_DIM + 255) / 256, 256, 0, stream>>>(out, M_TOK * K_DIM);
  zero_f32_kernel<<<1, 64, 0, stream>>>((float*)counts, 64);
  route_kernel<<<(CAP + 255) / 256, 256, 0, stream>>>(tki, tkw, counts, perm, wgt);

  for (int e = 0; e < N_EXP; ++e) {
    moe_gemm1<<<dim3(CAP / BM, N_INT / BN), 256, 0, stream>>>(
        x, gup + (size_t)e * 2 * N_INT * K_DIM, counts, perm + e * CAP, H, e);
    moe_gemm2<<<dim3(CAP / BM, K_DIM / BN), 256, 0, stream>>>(
        H, dpj + (size_t)e * K_DIM * N_INT, counts, perm + e * CAP, wgt + e * CAP, out, e);
  }
}
